// Att_0_layer1_11879879544264
// MI455X (gfx1250) — matrix operand order repaired and checked by reading, not compiled, not run on hardware
//
#include <hip/hip_runtime.h>
#include <hip/hip_bf16.h>

typedef __attribute__((ext_vector_type(16))) _Float16 v16h;
typedef __attribute__((ext_vector_type(8)))  _Float16 v8h;
typedef __attribute__((ext_vector_type(8)))  float    v8f;

#define KB_  36
#define VD   1024
#define QD   512
#define HD   512
#define BD   96
#define SD   4
#define TD   16
#define NN   3264              // fixed by reference setup (24 * 136)
#define NR   (NN * KB_)        // 117504 rows of the big GEMM

// workspace layout (bytes)
#define OFF_WPV   0u                         // packed f16 w1[:1024]  : 32*32 tiles * 512 halves = 1 MB
#define OFF_WPQ   (1048576u)                 // packed f16 w1[1024:]  : 16*32 tiles * 512 halves = 512 KB
#define OFF_QPROJ (OFF_WPQ + 524288u)        // qproj: NN*512 f32 = 6684672 B
#define OFF_LOGIT (OFF_QPROJ + 6684672u)     // logits: NR f32 = 470016 B
#define OFF_LEN   (OFF_LOGIT + 470016u)      // 384 ints
#define OFF_START (OFF_LEN + 1536u)          // 384 ints

// ---------------------------------------------------------------------------
// Pack w1 (1536x512 f32, row-major) into WMMA B-fragment lane layout (f16).
// Tile = 32(k) x 16(n). Lane L: col n = L&15, kbase = (L<16?0:8),
// element e -> W[kt*32 + kbase + e + (e>=8?8:0)][ct*16 + n]. Lane data contiguous (32 B).
// ---------------------------------------------------------------------------
__global__ void pack_w1_kernel(const float* __restrict__ w1,
                               _Float16* __restrict__ wpv,
                               _Float16* __restrict__ wpq) {
  int wave = (blockIdx.x * blockDim.x + threadIdx.x) >> 5;
  int L = threadIdx.x & 31;
  if (wave >= 1536) return;                 // 1024 v-tiles + 512 q-tiles
  int n  = L & 15;
  int kb = (L < 16) ? 0 : 8;
  const float* src;
  _Float16* dst;
  int kt, ct;
  if (wave < 1024) {
    kt = wave >> 5; ct = wave & 31;
    src = w1;                dst = wpv + (size_t)wave * 512;
  } else {
    int qi = wave - 1024;
    kt = qi >> 5; ct = qi & 31;
    src = w1 + 1024 * HD;    dst = wpq + (size_t)qi * 512;
  }
  v16h frag;
#pragma unroll
  for (int e = 0; e < 16; ++e)
    frag[e] = (_Float16)src[(size_t)(kt * 32 + kb + e + (e >= 8 ? 8 : 0)) * HD + ct * 16 + n];
  *(v16h*)(dst + L * 16) = frag;
}

// ---------------------------------------------------------------------------
// Sequence lengths + exclusive prefix (starts) from tags_attention (B,S,T).
// ---------------------------------------------------------------------------
__global__ void lens_kernel(const int* __restrict__ tags,
                            int* __restrict__ lens, int* __restrict__ starts) {
  __shared__ int sl[BD * SD];
  int j = threadIdx.x;
  if (j < BD * SD) {
    int s = 0;
#pragma unroll
    for (int t = 0; t < TD; ++t) s += tags[j * TD + t];
    sl[j] = s;
    lens[j] = s;
  }
  __syncthreads();
  if (j == 0) {
    int run = 0;
    for (int i = 0; i < BD * SD; ++i) { starts[i] = run; run += sl[i]; }
  }
}

// ---------------------------------------------------------------------------
// qproj[n,h] = q[n,:] @ w1[1024:,h] + b1[h]   (NN x 512, K=512)
// One workgroup = 16 rows; 8 waves each own 4 column tiles (64 cols).
// Double-buffered LDS staging + register prefetch of next fp32 A chunk.
// ---------------------------------------------------------------------------
__global__ __launch_bounds__(256) void qproj_kernel(
    const float* __restrict__ q, const _Float16* __restrict__ wpq,
    const float* __restrict__ b1, float* __restrict__ qproj) {
  __shared__ __align__(32) _Float16 ldsA[2 * 512];
  int tau = threadIdx.x;
  int w = tau >> 5, L = tau & 31;
  int rbase = blockIdx.x * 16;
  v8f acc[4];
#pragma unroll
  for (int jj = 0; jj < 4; ++jj)
#pragma unroll
    for (int e = 0; e < 8; ++e) acc[jj][e] = 0.0f;

  bool filler = (tau < 64);
  int Lf = (tau & 63) >> 1, g = tau & 1;
  int mrow = Lf & 15, klane = (Lf < 16) ? 0 : 8;
  const float* srcbase = q + (size_t)(rbase + mrow) * QD + klane + g * 16;
  _Float16* ldst = &ldsA[Lf * 16 + g * 8];

  const int NK = QD / 32;
  float4 pf0, pf1;
  if (filler) {
    pf0 = *(const float4*)(srcbase);
    pf1 = *(const float4*)(srcbase + 4);
  }
  for (int kt = 0; kt < NK; ++kt) {
    if (filler) {
      v8h hh;
      hh[0] = (_Float16)pf0.x; hh[1] = (_Float16)pf0.y; hh[2] = (_Float16)pf0.z; hh[3] = (_Float16)pf0.w;
      hh[4] = (_Float16)pf1.x; hh[5] = (_Float16)pf1.y; hh[6] = (_Float16)pf1.z; hh[7] = (_Float16)pf1.w;
      *(v8h*)(ldst + (kt & 1) * 512) = hh;
      int ktn = (kt + 1 < NK) ? kt + 1 : kt;     // prefetch next chunk
      pf0 = *(const float4*)(srcbase + ktn * 32);
      pf1 = *(const float4*)(srcbase + ktn * 32 + 4);
    }
    v16h b[4];                                   // issue B loads pre-barrier (L2 hot)
#pragma unroll
    for (int jj = 0; jj < 4; ++jj)
      b[jj] = *(const v16h*)(wpq + ((size_t)(kt * 32 + (w * 4 + jj)) * 512 + L * 16));
    __syncthreads();
    v16h a = *(const v16h*)(&ldsA[(kt & 1) * 512 + L * 16]);
#pragma unroll
    for (int jj = 0; jj < 4; ++jj)
      acc[jj] = __builtin_amdgcn_wmma_f32_16x16x32_f16(false, a, false, b[jj],
                                                       (short)0, acc[jj], false, false);
  }
  int n = L & 15, hi8 = (L >= 16) ? 8 : 0;
#pragma unroll
  for (int jj = 0; jj < 4; ++jj) {
    int h = (w * 4 + jj) * 16 + n;
    float bb = b1[h];
#pragma unroll
    for (int i = 0; i < 8; ++i)
      qproj[(size_t)(rbase + i + hi8) * HD + h] = acc[jj][i] + bb;
  }
}

// ---------------------------------------------------------------------------
// Main fused kernel: for 64 rows r (r = n*36+k):
//   joint[r,h] = relu( v[r,:] @ w1v[:,h] + qproj[r/36,h] )
//   logits[r]  = sum_h joint[r,h] * w2[h]
// 8 waves: wave w owns cols [w*64, w*64+64) (4 tiles), 4 row tiles each.
// Double-buffered LDS A staging, register prefetch, one barrier per k-step.
// ---------------------------------------------------------------------------
__global__ __launch_bounds__(256, 1) void fused_kernel(
    const float* __restrict__ v, const _Float16* __restrict__ wpv,
    const float* __restrict__ qproj, const float* __restrict__ w2f,
    float* __restrict__ logits) {
  __shared__ __align__(32) _Float16 ldsA[2 * 2048];
  __shared__ float ldsP[8 * 64];
  int tau = threadIdx.x;
  int w = tau >> 5, L = tau & 31;
  int rbase = blockIdx.x * 64;

  v8f acc[4][4];
#pragma unroll
  for (int rt = 0; rt < 4; ++rt)
#pragma unroll
    for (int jj = 0; jj < 4; ++jj)
#pragma unroll
      for (int e = 0; e < 8; ++e) acc[rt][jj][e] = 0.0f;

  // cooperative A-staging assignment (256 threads -> 2048 halves/step)
  int frt = tau >> 6;
  int fj  = tau & 63;
  int Lf = fj >> 1, g = fj & 1;
  int mrow = Lf & 15, klane = (Lf < 16) ? 0 : 8;
  const float* srcbase =
      v + (size_t)(rbase + frt * 16 + mrow) * VD + klane + g * 16;
  _Float16* ldst = &ldsA[frt * 512 + Lf * 16 + g * 8];

  const int NK = VD / 32;
  float4 pf0 = *(const float4*)(srcbase);
  float4 pf1 = *(const float4*)(srcbase + 4);

  for (int kt = 0; kt < NK; ++kt) {
    {                                             // store staged chunk (regs -> LDS f16)
      v8h hh;
      hh[0] = (_Float16)pf0.x; hh[1] = (_Float16)pf0.y; hh[2] = (_Float16)pf0.z; hh[3] = (_Float16)pf0.w;
      hh[4] = (_Float16)pf1.x; hh[5] = (_Float16)pf1.y; hh[6] = (_Float16)pf1.z; hh[7] = (_Float16)pf1.w;
      *(v8h*)(ldst + (kt & 1) * 2048) = hh;
    }
    {                                             // prefetch next A chunk into regs
      int ktn = (kt + 1 < NK) ? kt + 1 : kt;
      pf0 = *(const float4*)(srcbase + ktn * 32);
      pf1 = *(const float4*)(srcbase + ktn * 32 + 4);
    }
    v16h b[4];                                    // B loads issued pre-barrier (L2 hot)
#pragma unroll
    for (int jj = 0; jj < 4; ++jj)
      b[jj] = *(const v16h*)(wpv + ((size_t)(kt * 32 + (w * 4 + jj)) * 512 + L * 16));
    __syncthreads();
    v16h a[4];
#pragma unroll
    for (int rt = 0; rt < 4; ++rt)
      a[rt] = *(const v16h*)(&ldsA[(kt & 1) * 2048 + rt * 512 + L * 16]);
#pragma unroll
    for (int rt = 0; rt < 4; ++rt)
#pragma unroll
      for (int jj = 0; jj < 4; ++jj)
        acc[rt][jj] = __builtin_amdgcn_wmma_f32_16x16x32_f16(
            false, a[rt], false, b[jj], (short)0, acc[rt][jj], false, false);
  }

  // epilogue: + qproj, relu, * w2, reduce over h
  int n = L & 15, hi8 = (L >= 16) ? 8 : 0;
  float srow[4][8];
#pragma unroll
  for (int rt = 0; rt < 4; ++rt)
#pragma unroll
    for (int i = 0; i < 8; ++i) srow[rt][i] = 0.0f;

#pragma unroll
  for (int rt = 0; rt < 4; ++rt) {
#pragma unroll
    for (int jj = 0; jj < 4; ++jj) {
      int h = (w * 4 + jj) * 16 + n;
      float w2h = w2f[h];
#pragma unroll
      for (int i = 0; i < 8; ++i) {
        int r = rbase + rt * 16 + i + hi8;
        int nrow = r / KB_;
        float val = acc[rt][jj][i] + qproj[(size_t)nrow * HD + h];
        val = fmaxf(val, 0.0f);
        srow[rt][i] += val * w2h;
      }
    }
  }
  // butterfly reduce within each 16-lane half (cols of this wave's h-range)
#pragma unroll
  for (int off = 1; off <= 8; off <<= 1)
#pragma unroll
    for (int rt = 0; rt < 4; ++rt)
#pragma unroll
      for (int i = 0; i < 8; ++i)
        srow[rt][i] += __shfl_xor(srow[rt][i], off, 32);

  if (n == 0) {                              // lanes 0 and 16 hold full half-sums
#pragma unroll
    for (int rt = 0; rt < 4; ++rt)
#pragma unroll
      for (int i = 0; i < 8; ++i)
        ldsP[w * 64 + rt * 16 + i + hi8] = srow[rt][i];
  }
  __syncthreads();
  if (tau < 64) {                            // deterministic cross-wave sum
    float s = 0.0f;
#pragma unroll
    for (int wv = 0; wv < 8; ++wv) s += ldsP[wv * 64 + tau];
    logits[rbase + tau] = s;
  }
}

// ---------------------------------------------------------------------------
// Gather + masked softmax + renormalize. One thread per (b,s,t) row.
// ---------------------------------------------------------------------------
__global__ void softmax_kernel(const float* __restrict__ logits,
                               const float* __restrict__ box_mask,
                               const float* __restrict__ b2p,
                               const int* __restrict__ lens,
                               const int* __restrict__ starts,
                               float* __restrict__ out) {
  int p = blockIdx.x * blockDim.x + threadIdx.x;
  if (p >= BD * SD * TD) return;
  int b = p / (SD * TD);
  int rem = p % (SD * TD);
  int t = rem % TD;
  int j = b * SD + rem / TD;
  int len = lens[j];
  bool valid = t < len;
  int nrow = valid ? (starts[j] + t) : 0;
  float b2 = b2p[0];
  const float* mrow = box_mask + b * KB_;
  const float* lrow = logits + (size_t)nrow * KB_;

  float mx = -1e30f;
  for (int k = 0; k < KB_; ++k) {
    float x = valid ? (lrow[k] + b2) : 0.0f;
    x *= mrow[k];
    mx = fmaxf(mx, x);
  }
  float sumE = 0.0f;
  for (int k = 0; k < KB_; ++k) {
    float x = valid ? (lrow[k] + b2) : 0.0f;
    sumE += expf(x * mrow[k] - mx);
  }
  float invE = 1.0f / sumE;
  float sum2 = 0.0f;
  for (int k = 0; k < KB_; ++k) {
    float x = valid ? (lrow[k] + b2) : 0.0f;
    sum2 += expf(x * mrow[k] - mx) * invE * mrow[k];
  }
  float inv = 1.0f / (sum2 + 1e-13f);
  for (int k = 0; k < KB_; ++k) {
    float x = valid ? (lrow[k] + b2) : 0.0f;
    out[(size_t)p * KB_ + k] = expf(x * mrow[k] - mx) * invE * mrow[k] * inv;
  }
}

extern "C" void kernel_launch(void* const* d_in, const int* in_sizes, int n_in,
                              void* d_out, int out_size, void* d_ws, size_t ws_size,
                              hipStream_t stream) {
  const float* v        = (const float*)d_in[0];
  const float* q        = (const float*)d_in[1];
  const float* box_mask = (const float*)d_in[2];
  const int*   tags     = (const int*)d_in[3];
  const float* w1       = (const float*)d_in[4];
  const float* b1       = (const float*)d_in[5];
  const float* w2       = (const float*)d_in[6];
  const float* b2       = (const float*)d_in[7];
  float* out = (float*)d_out;

  char* ws = (char*)d_ws;
  _Float16* wpv   = (_Float16*)(ws + OFF_WPV);
  _Float16* wpq   = (_Float16*)(ws + OFF_WPQ);
  float*    qproj = (float*)(ws + OFF_QPROJ);
  float*    logit = (float*)(ws + OFF_LOGIT);
  int*      lens  = (int*)(ws + OFF_LEN);
  int*      strt  = (int*)(ws + OFF_START);

  pack_w1_kernel<<<384, 128, 0, stream>>>(w1, wpv, wpq);
  lens_kernel<<<1, 384, 0, stream>>>(tags, lens, strt);
  qproj_kernel<<<NN / 16, 256, 0, stream>>>(q, wpq, b1, qproj);
  fused_kernel<<<NR / 64, 256, 0, stream>>>(v, wpv, qproj, w2, logit);
  softmax_kernel<<<(BD * SD * TD + 255) / 256, 256, 0, stream>>>(
      logit, box_mask, b2, lens, strt, out);
}
